// SwinTransformerBlock_22247930593459
// MI455X (gfx1250) — matrix operand order repaired and checked by reading, not compiled, not run on hardware
//
#include <hip/hip_runtime.h>
#include <hip/hip_bf16.h>

typedef __attribute__((ext_vector_type(16))) _Float16 v16h;
typedef __attribute__((ext_vector_type(8)))  _Float16 v8h;
typedef __attribute__((ext_vector_type(8)))  float    v8f;

#define RES_    56
#define WS_     7
#define SHIFT_  3
#define DIM_    192
#define HEADS_  6
#define HD_     32
#define NTOK_   49
#define BATCH_  32
#define NWIN_   64                 // 8x8 windows per image
#define BW_     (BATCH_*NWIN_)     // 2048 windows
#define MROWS_  (BW_*NTOK_)        // 100352 rows (== B*L)
#define SCALE_  0.17677669529663687f
#define LN_EPS_ 1e-5f

// ---------------- workspace layout (bytes) ----------------
#define SZ_H    ((size_t)MROWS_*DIM_*2)        // h / ln2-out, f16
#define SZ_QKV  ((size_t)MROWS_*768*2)         // qkv / fc1-act, f16
#define SZ_ATT  ((size_t)MROWS_*DIM_*2)        // attn out f16
#define SZ_X2   ((size_t)MROWS_*DIM_*4)        // x2 f32
#define OFF_H    ((size_t)0)
#define OFF_QKV  (OFF_H   + SZ_H)
#define OFF_ATT  (OFF_QKV + SZ_QKV)
#define OFF_X2   (OFF_ATT + SZ_ATT)
#define OFF_WQ   (OFF_X2  + SZ_X2)
#define OFF_WP   (OFF_WQ  + (size_t)192*576*2)
#define OFF_W1   (OFF_WP  + (size_t)192*192*2)
#define OFF_W2   (OFF_W1  + (size_t)192*768*2)
#define OFF_BM   (OFF_W2  + (size_t)768*192*2) // bias+mask table [4][6][64][64] f32

__device__ __forceinline__ float wave_reduce_sum(float v) {
    v += __shfl_xor(v, 16, 32);
    v += __shfl_xor(v,  8, 32);
    v += __shfl_xor(v,  4, 32);
    v += __shfl_xor(v,  2, 32);
    v += __shfl_xor(v,  1, 32);
    return v;
}

// ---------------- weight convert + transpose: Wt[n][k] = W[k][n] ----------------
__global__ void convt_kernel(const float* __restrict__ W, _Float16* __restrict__ Wt,
                             int K, int Ncol) {
    int idx = blockIdx.x * 256 + threadIdx.x;
    if (idx >= K * Ncol) return;
    int n = idx / K, k = idx - n * K;
    Wt[idx] = (_Float16)W[k * Ncol + n];
}

// ---------------- precompute padded bias+mask table BM[4][6][64][64] ----------------
__device__ __forceinline__ int region_(int p) {
    return (p < RES_ - WS_) ? 0 : ((p < RES_ - SHIFT_) ? 1 : 2);
}
__global__ void bm_kernel(const float* __restrict__ btab, float* __restrict__ bm) {
    int idx = blockIdx.x * 256 + threadIdx.x;
    if (idx >= 4 * HEADS_ * 64 * 64) return;
    int mcol = idx & 63;
    int nrow = (idx >> 6) & 63;
    int ch   = idx >> 12;             // cls*6 + head
    int head = ch % HEADS_;
    int cls  = ch / HEADS_;
    float val;
    if (mcol >= NTOK_)      val = -1e30f;   // padded key columns -> softmax 0
    else if (nrow >= NTOK_) val = 0.f;      // padded query rows (unused output)
    else {
        int i0 = nrow / WS_, i1 = nrow - i0 * WS_;
        int j0 = mcol / WS_, j1 = mcol - j0 * WS_;
        int rel = (i0 - j0 + WS_ - 1) * (2 * WS_ - 1) + (i1 - j1 + WS_ - 1);
        float bv = btab[rel * HEADS_ + head];
        int wh = (cls & 2) ? 7 : 0;         // representative window row/col
        int ww = (cls & 1) ? 7 : 0;
        int cntn = region_(wh * WS_ + i0) * 3 + region_(ww * WS_ + i1);
        int cntm = region_(wh * WS_ + j0) * 3 + region_(ww * WS_ + j1);
        val = bv + ((cntn != cntm) ? -100.f : 0.f);
    }
    bm[idx] = val;
}

// ---------------- LN1 + shift + window partition (1 token per wave) ----------------
__global__ __launch_bounds__(256)
void ln_window_kernel(const float* __restrict__ x, const float* __restrict__ g,
                      const float* __restrict__ bb, _Float16* __restrict__ h) {
    const int t  = blockIdx.x * 8 + (threadIdx.x >> 5);   // dest window-token row
    const int lane = threadIdx.x & 31;
    const int bw = t / NTOK_, n = t - bw * NTOK_;
    const int b  = bw >> 6, w = bw & 63;
    const int wh = w >> 3, ww = w & 7;
    const int i  = n / WS_, j = n - i * WS_;
    int R = wh * WS_ + i + SHIFT_; if (R >= RES_) R -= RES_;
    int C = ww * WS_ + j + SHIFT_; if (C >= RES_) C -= RES_;
    const float* src = x + ((size_t)b * (RES_ * RES_) + R * RES_ + C) * DIM_;
    float v[6]; float s = 0.f;
#pragma unroll
    for (int k = 0; k < 6; ++k) { v[k] = src[lane + k * 32]; s += v[k]; }
    s = wave_reduce_sum(s);
    const float mean = s * (1.0f / DIM_);
    float q = 0.f;
#pragma unroll
    for (int k = 0; k < 6; ++k) { float d = v[k] - mean; q += d * d; }
    q = wave_reduce_sum(q);
    const float rstd = rsqrtf(q * (1.0f / DIM_) + LN_EPS_);
    _Float16* dst = h + (size_t)t * DIM_;
#pragma unroll
    for (int k = 0; k < 6; ++k) {
        int c = lane + k * 32;
        dst[c] = (_Float16)((v[k] - mean) * rstd * g[c] + bb[c]);
    }
}

// ---------------- plain LN2 (1 token per wave) ----------------
__global__ __launch_bounds__(256)
void ln_plain_kernel(const float* __restrict__ x, const float* __restrict__ g,
                     const float* __restrict__ bb, _Float16* __restrict__ h) {
    const int t = blockIdx.x * 8 + (threadIdx.x >> 5);
    const int lane = threadIdx.x & 31;
    const float* src = x + (size_t)t * DIM_;
    float v[6]; float s = 0.f;
#pragma unroll
    for (int k = 0; k < 6; ++k) { v[k] = src[lane + k * 32]; s += v[k]; }
    s = wave_reduce_sum(s);
    const float mean = s * (1.0f / DIM_);
    float q = 0.f;
#pragma unroll
    for (int k = 0; k < 6; ++k) { float d = v[k] - mean; q += d * d; }
    q = wave_reduce_sum(q);
    const float rstd = rsqrtf(q * (1.0f / DIM_) + LN_EPS_);
    _Float16* dst = h + (size_t)t * DIM_;
#pragma unroll
    for (int k = 0; k < 6; ++k) {
        int c = lane + k * 32;
        dst[c] = (_Float16)((v[k] - mean) * rstd * g[c] + bb[c]);
    }
}

// ---------------- WMMA GEMM: C[M,N] = A[M,KT](f16) * Wt[N,KT](f16) + bias ----------------
// wave tile 32x64 (2 A-fragments x 4 B-fragments, 8 WMMA per k-step); KT compile-time
// MODE 0: f16 out (qkv)
// MODE 1: exact GELU, f16 out (fc1)
// MODE 2: window-reverse scatter + residual(x) -> f32 x2 (proj)
// MODE 3: row-direct residual(x2) -> f32 out (fc2)
template<int MODE, int KT>
__global__ __launch_bounds__(128)
void gemm_wmma_kernel(const _Float16* __restrict__ A, const _Float16* __restrict__ Bt,
                      const float* __restrict__ bias,
                      _Float16* __restrict__ outH, float* __restrict__ outF,
                      const float* __restrict__ resid,
                      int Ncol) {
    const int lane = threadIdx.x & 31;
    const int wave = threadIdx.x >> 5;
    const int half = lane >> 4;
    const int l15  = lane & 15;
    const int row0 = blockIdx.x * 128 + wave * 32;
    const int col0 = blockIdx.y * 64;

    v8f acc[2][4] = {};
    const _Float16* arow0 = A + (size_t)(row0 + l15) * KT;
    const _Float16* arow1 = A + (size_t)(row0 + 16 + l15) * KT;
    const _Float16* bcol  = Bt + (size_t)(col0 + l15) * KT + half * 16;

    for (int k0 = 0; k0 < KT; k0 += 32) {
        const int ka = k0 + half * 16;
        v8h a0lo = *(const v8h*)(arow0 + ka);
        v8h a0hi = *(const v8h*)(arow0 + ka + 8);
        v8h a1lo = *(const v8h*)(arow1 + ka);
        v8h a1hi = *(const v8h*)(arow1 + ka + 8);
        v16h a0, a1;
#pragma unroll
        for (int j = 0; j < 8; ++j) {
            a0[j] = a0lo[j]; a0[j + 8] = a0hi[j];
            a1[j] = a1lo[j]; a1[j + 8] = a1hi[j];
        }
#pragma unroll
        for (int t = 0; t < 4; ++t) {
            v16h bfr = *(const v16h*)(bcol + (size_t)t * 16 * KT + k0);
            acc[0][t] = __builtin_amdgcn_wmma_f32_16x16x32_f16(
                false, a0, false, bfr, (short)0, acc[0][t], false, false);
            acc[1][t] = __builtin_amdgcn_wmma_f32_16x16x32_f16(
                false, a1, false, bfr, (short)0, acc[1][t], false, false);
        }
    }

#pragma unroll
    for (int r = 0; r < 2; ++r) {
#pragma unroll
        for (int t = 0; t < 4; ++t) {
            const int col = col0 + t * 16 + l15;
            const float bv = bias[col];
#pragma unroll
            for (int i = 0; i < 8; ++i) {
                const int row = row0 + r * 16 + half * 8 + i;
                float v = acc[r][t][i] + bv;
                if (MODE == 0) {
                    outH[(size_t)row * Ncol + col] = (_Float16)v;
                } else if (MODE == 1) {
                    v = 0.5f * v * (1.0f + erff(v * 0.70710678118654752f));
                    outH[(size_t)row * Ncol + col] = (_Float16)v;
                } else if (MODE == 2) {
                    int bw = row / NTOK_, n = row - bw * NTOK_;
                    int b = bw >> 6, w = bw & 63;
                    int wh = w >> 3, ww = w & 7;
                    int i0 = n / WS_, j0 = n - i0 * WS_;
                    int R = wh * WS_ + i0 + SHIFT_; if (R >= RES_) R -= RES_;
                    int C = ww * WS_ + j0 + SHIFT_; if (C >= RES_) C -= RES_;
                    size_t idx = ((size_t)b * (RES_ * RES_) + R * RES_ + C) * DIM_ + col;
                    outF[idx] = resid[idx] + v;
                } else {
                    size_t idx = (size_t)row * Ncol + col;
                    outF[idx] = resid[idx] + v;
                }
            }
        }
    }
}

// ---------------- attention: one block per (window, head), 4 waves ----------------
__global__ __launch_bounds__(128)
void attn_wmma_kernel(const _Float16* __restrict__ qkv,
                      const float* __restrict__ bm,
                      _Float16* __restrict__ out) {
    __shared__ __align__(32) _Float16 Qs[64][32];
    __shared__ __align__(32) _Float16 Ks[64][32];
    __shared__ __align__(32) _Float16 Vs[64][32];
    __shared__ __align__(32) float    Ss[64][64];
    __shared__ __align__(32) _Float16 Ps[64][64];

    const int bw   = blockIdx.x / HEADS_;
    const int head = blockIdx.x - bw * HEADS_;
    const int tid  = threadIdx.x;
    const int lane = tid & 31, wave = tid >> 5;
    const int half = lane >> 4, l15 = lane & 15;
    const size_t rowbase = (size_t)bw * NTOK_;

    // stage Q,K,V (49x32 padded to 64x32) into LDS with 16-byte vectors
    const v8h zero8 = {};
#pragma unroll
    for (int it = 0; it < 2; ++it) {
        int idx = tid + it * 128;          // 256 chunks of 8 f16 per matrix
        int r = idx >> 2, ch = (idx & 3) * 8;
        v8h q = zero8, k = zero8, v = zero8;
        if (r < NTOK_) {
            const _Float16* p = qkv + (rowbase + r) * (3 * DIM_) + head * HD_ + ch;
            q = *(const v8h*)(p);
            k = *(const v8h*)(p + DIM_);
            v = *(const v8h*)(p + 2 * DIM_);
        }
        *(v8h*)&Qs[r][ch] = q;
        *(v8h*)&Ks[r][ch] = k;
        *(v8h*)&Vs[r][ch] = v;
    }
    __syncthreads();

    const int band = wave * 16;
    const int w = bw & 63;
    const int cls = (((w >> 3) == 7) ? 2 : 0) | (((w & 7) == 7) ? 1 : 0);
    const float* bmp = bm + ((size_t)(cls * HEADS_ + head) << 12);   // [64][64]

    // ---- S = (Q K^T) * scale + (bias+mask)[padded] ----
    {
        const int mrow = band + l15;
        v8h alo = *(const v8h*)&Qs[mrow][half * 16];
        v8h ahi = *(const v8h*)&Qs[mrow][half * 16 + 8];
        v16h a;
#pragma unroll
        for (int j = 0; j < 8; ++j) { a[j] = alo[j]; a[j + 8] = ahi[j]; }
        v8f cz = {};
#pragma unroll
        for (int t = 0; t < 4; ++t) {
            const int mcol = t * 16 + l15;
            v16h bfr = *(const v16h*)&Ks[mcol][half * 16];
            v8f s = __builtin_amdgcn_wmma_f32_16x16x32_f16(
                false, a, false, bfr, (short)0, cz, false, false);
#pragma unroll
            for (int i = 0; i < 8; ++i) {
                const int nrow = band + half * 8 + i;
                Ss[nrow][mcol] = s[i] * SCALE_ + bmp[nrow * 64 + mcol];
            }
        }
    }
    __syncthreads();

    // ---- softmax: 2 lanes per row, 32 cols each ----
    {
        const int r  = band + l15;
        const int cb = half * 32;
        float ev[32];
        float mx = -1e30f;
#pragma unroll
        for (int c = 0; c < 32; ++c) { ev[c] = Ss[r][cb + c]; mx = fmaxf(mx, ev[c]); }
        mx = fmaxf(mx, __shfl_xor(mx, 16, 32));
        float sum = 0.f;
#pragma unroll
        for (int c = 0; c < 32; ++c) { ev[c] = __expf(ev[c] - mx); sum += ev[c]; }
        sum += __shfl_xor(sum, 16, 32);
        const float inv = 1.0f / sum;
#pragma unroll
        for (int c = 0; c < 32; ++c) Ps[r][cb + c] = (_Float16)(ev[c] * inv);
    }
    __syncthreads();

    // ---- O = P V ----
    v8f acc0 = {}, acc1 = {};
#pragma unroll
    for (int s0 = 0; s0 < 64; s0 += 32) {
        const int mrow = band + l15;
        v8h alo = *(const v8h*)&Ps[mrow][s0 + half * 16];
        v8h ahi = *(const v8h*)&Ps[mrow][s0 + half * 16 + 8];
        v16h a;
#pragma unroll
        for (int j = 0; j < 8; ++j) { a[j] = alo[j]; a[j + 8] = ahi[j]; }
        v16h b0, b1;
#pragma unroll
        for (int j = 0; j < 16; ++j) {
            b0[j] = Vs[s0 + half * 16 + j][l15];
            b1[j] = Vs[s0 + half * 16 + j][16 + l15];
        }
        acc0 = __builtin_amdgcn_wmma_f32_16x16x32_f16(false, a, false, b0, (short)0, acc0, false, false);
        acc1 = __builtin_amdgcn_wmma_f32_16x16x32_f16(false, a, false, b1, (short)0, acc1, false, false);
    }
#pragma unroll
    for (int i = 0; i < 8; ++i) {
        int nrow = band + half * 8 + i;
        if (nrow < NTOK_) {
            _Float16* dst = out + (rowbase + nrow) * DIM_ + head * HD_;
            dst[l15]      = (_Float16)acc0[i];
            dst[16 + l15] = (_Float16)acc1[i];
        }
    }
}

extern "C" void kernel_launch(void* const* d_in, const int* in_sizes, int n_in,
                              void* d_out, int out_size, void* d_ws, size_t ws_size,
                              hipStream_t stream) {
    (void)in_sizes; (void)n_in; (void)out_size; (void)ws_size;
    const float* x      = (const float*)d_in[0];
    const float* n1g    = (const float*)d_in[1];
    const float* n1b    = (const float*)d_in[2];
    const float* qkv_w  = (const float*)d_in[3];
    const float* qkv_b  = (const float*)d_in[4];
    const float* proj_w = (const float*)d_in[5];
    const float* proj_b = (const float*)d_in[6];
    const float* btab   = (const float*)d_in[7];
    const float* n2g    = (const float*)d_in[8];
    const float* n2b    = (const float*)d_in[9];
    const float* fc1_w  = (const float*)d_in[10];
    const float* fc1_b  = (const float*)d_in[11];
    const float* fc2_w  = (const float*)d_in[12];
    const float* fc2_b  = (const float*)d_in[13];
    float* out = (float*)d_out;
    char*  ws  = (char*)d_ws;

    _Float16* hbuf = (_Float16*)(ws + OFF_H);    // LN1 windows / LN2 out
    _Float16* qkvb = (_Float16*)(ws + OFF_QKV);  // qkv / fc1 activations
    _Float16* attb = (_Float16*)(ws + OFF_ATT);  // attention output
    float*    x2   = (float*)   (ws + OFF_X2);   // residual mid-point
    _Float16* wq   = (_Float16*)(ws + OFF_WQ);
    _Float16* wp   = (_Float16*)(ws + OFF_WP);
    _Float16* w1   = (_Float16*)(ws + OFF_W1);
    _Float16* w2   = (_Float16*)(ws + OFF_W2);
    float*    bm   = (float*)   (ws + OFF_BM);

    // weights -> f16 transposed [N][K]; bias+mask table
    convt_kernel<<<(192*576 + 255)/256, 256, 0, stream>>>(qkv_w, wq, 192, 576);
    convt_kernel<<<(192*192 + 255)/256, 256, 0, stream>>>(proj_w, wp, 192, 192);
    convt_kernel<<<(192*768 + 255)/256, 256, 0, stream>>>(fc1_w, w1, 192, 768);
    convt_kernel<<<(768*192 + 255)/256, 256, 0, stream>>>(fc2_w, w2, 768, 192);
    bm_kernel<<<(4*HEADS_*64*64 + 255)/256, 256, 0, stream>>>(btab, bm);

    // LN1 + shift + window partition
    ln_window_kernel<<<MROWS_/8, 256, 0, stream>>>(x, n1g, n1b, hbuf);

    // QKV GEMM (100352 x 576 x 192)
    gemm_wmma_kernel<0, 192><<<dim3(MROWS_/128, 576/64), 128, 0, stream>>>(
        hbuf, wq, qkv_b, qkvb, nullptr, nullptr, 576);

    // attention per (window, head)
    attn_wmma_kernel<<<BW_*HEADS_, 128, 0, stream>>>(qkvb, bm, attb);

    // proj GEMM + window-reverse scatter + residual(x) -> x2
    gemm_wmma_kernel<2, 192><<<dim3(MROWS_/128, 192/64), 128, 0, stream>>>(
        attb, wp, proj_b, nullptr, x2, x, 192);

    // LN2
    ln_plain_kernel<<<MROWS_/8, 256, 0, stream>>>(x2, n2g, n2b, hbuf);

    // fc1 GEMM + exact GELU (100352 x 768 x 192)
    gemm_wmma_kernel<1, 192><<<dim3(MROWS_/128, 768/64), 128, 0, stream>>>(
        hbuf, w1, fc1_b, qkvb, nullptr, nullptr, 768);

    // fc2 GEMM + residual(x2) -> out (100352 x 192 x 768)
    gemm_wmma_kernel<3, 768><<<dim3(MROWS_/128, 192/64), 128, 0, stream>>>(
        qkvb, w2, fc2_b, nullptr, out, x2, 192);
}
